// SC_block_38809324487318
// MI455X (gfx1250) — matrix operand order repaired and checked by reading, not compiled, not run on hardware
//
#include <hip/hip_runtime.h>
#include <hip/hip_bf16.h>
#include <math.h>

typedef _Float16 half_t;
typedef __attribute__((ext_vector_type(16))) _Float16 v16h;
typedef __attribute__((ext_vector_type(8)))  _Float16 v8h;
typedef __attribute__((ext_vector_type(8)))  float    v8f;
typedef __attribute__((ext_vector_type(4)))  unsigned int v4u;
typedef __attribute__((ext_vector_type(8)))  int v8i;
typedef __attribute__((ext_vector_type(4)))  int v4i;

#define CC     512
#define KK     256
#define HWSZ   25600
#define WD     160
#define CATC   1024
#define EPSV   1e-5f

// GEMM tiling: 128x128 block tile, K-step 32, 8 waves, wave tile 32x64 (8 WMMA/step)
#define BM 128
#define BN 128
#define BK 32
#define LDA 40   // halfs per row: 80B stride -> conflict-free & 16B aligned (TDM pads to this)
#define LDB 40

#if __has_builtin(__builtin_amdgcn_tensor_load_to_lds) && \
    __has_builtin(__builtin_amdgcn_s_wait_tensorcnt)
#define HAVE_TDM 1
#endif

union Frag { v16h v; v8h h[2]; };

// ---- CDNA5 async memory->LDS (GLOBAL_LOAD_ASYNC_TO_LDS_B128, ASYNCcnt tracked) ----
__device__ __forceinline__ void async_b128(const half_t* g, half_t* l) {
  auto gp = (const __attribute__((address_space(1))) half_t*)g;
  auto lp = (__attribute__((address_space(3))) half_t*)l;
  asm volatile("global_load_async_to_lds_b128 %0, %1, off" :: "v"(lp), "v"(gp) : "memory");
}
__device__ __forceinline__ void async_wait0() {
  asm volatile("s_wait_asynccnt 0" ::: "memory");
}

#ifdef HAVE_TDM
__device__ __forceinline__ unsigned lds_off(const half_t* p) {
  return (unsigned)(unsigned long long)(const __attribute__((address_space(3))) half_t*)p;
}
// TDM 2D tile load: tile = 32 halfs (dim0, contiguous) x tileD1 lines, line stride stride0
// (data_size units). LDS dest padded +4 DWORDs after every 16 DWORDs -> 40-half line pitch.
// D# per ISA ch.8: group0 {count|flags, lds_addr, gaddr_lo, gaddr_hi|type=2},
// group1 {mask/ds/pad, tdim0, tdim0_hi|tdim1, tdim1_hi|tile0, tile1, stride0_lo, ...}.
__device__ __forceinline__ void tdm_load_tile(const half_t* gsrc, unsigned ldsByte,
                                              unsigned tdim0, unsigned tdim1,
                                              unsigned stride0, unsigned tileD1) {
  unsigned long long ga = (unsigned long long)(size_t)gsrc;
  v4u g0;
  g0[0] = 1u;                                            // count=1, user mode
  g0[1] = ldsByte;                                       // lds_addr
  g0[2] = (unsigned)ga;                                  // global_addr[31:0]
  g0[3] = (unsigned)((ga >> 32) & 0x1FFFFFFu) | (2u << 30); // global_addr[56:32], type=2
  v8i g1;
  g1[0] = (int)((1u << 16)      // data_size = 2 bytes
              | (1u << 20)      // pad_enable
              | (3u << 22)      // pad_interval: 16 DWORDs
              | (3u << 25));    // pad_amount: 4 DWORDs
  g1[1] = (int)(tdim0 << 16);                            // tensor_dim0[15:0] @ [63:48]
  g1[2] = (int)((tdim0 >> 16) | (tdim1 << 16));          // tdim0[31:16] | tdim1[15:0]
  g1[3] = (int)((tdim1 >> 16) | (32u << 16));            // tdim1[31:16] | tile_dim0=32
  g1[4] = (int)tileD1;                                   // tile_dim1 (tile_dim2=0)
  g1[5] = (int)stride0;                                  // tensor_dim0_stride[31:0]
  g1[6] = 0;
  g1[7] = 0;
  v4i z4 = {0, 0, 0, 0};
#if defined(__clang_major__) && __clang_major__ >= 23
  v8i z8 = {0, 0, 0, 0, 0, 0, 0, 0};
  __builtin_amdgcn_tensor_load_to_lds(g0, g1, z4, z4, z8, 0);
#else
  __builtin_amdgcn_tensor_load_to_lds(g0, g1, z4, z4, 0);
#endif
}
#endif  // HAVE_TDM

// ---------------- per-channel mean / max ----------------
__global__ void k_reduce(const float* __restrict__ x, float* __restrict__ avg,
                         float* __restrict__ mx) {
  __shared__ float ssum[256];
  __shared__ float smax[256];
  int c = blockIdx.x, t = threadIdx.x;
  const float* p = x + (size_t)c * HWSZ;
  float s = 0.f, m = -INFINITY;
  for (int i = t; i < HWSZ; i += 256) { float v = p[i]; s += v; m = fmaxf(m, v); }
  ssum[t] = s; smax[t] = m; __syncthreads();
  for (int off = 128; off > 0; off >>= 1) {
    if (t < off) { ssum[t] += ssum[t + off]; smax[t] = fmaxf(smax[t], smax[t + off]); }
    __syncthreads();
  }
  if (t == 0) { avg[c] = ssum[0] * (1.0f / HWSZ); mx[c] = smax[0]; }
}

// ---------------- SE-MLP + sigmoid + sorted top-K via ranking ----------------
__global__ void k_mlp_topk(const float* __restrict__ avg, const float* __restrict__ mx,
                           const float* __restrict__ w_fc1, const float* __restrict__ w_fc2,
                           float* __restrict__ scales, int* __restrict__ idx) {
  __shared__ float sa[CC], sm[CC], hsum[KK], ssc[CC];
  __shared__ int sel[CC];
  int t = threadIdx.x;
  if (t < CC) { sa[t] = avg[t]; sm[t] = mx[t]; }
  __syncthreads();
  if (t < KK) {
    const float* wr = w_fc1 + (size_t)t * CC;
    float da = 0.f, dm = 0.f;
    for (int c = 0; c < CC; ++c) { float w = wr[c]; da += w * sa[c]; dm += w * sm[c]; }
    hsum[t] = fmaxf(da, 0.f) + fmaxf(dm, 0.f);
  }
  __syncthreads();
  if (t < CC) {
    const float* wr = w_fc2 + (size_t)t * KK;
    float o = 0.f;
    for (int k = 0; k < KK; ++k) o += wr[k] * hsum[k];
    float s = 1.f / (1.f + __expf(-o));
    ssc[t] = s; scales[t] = s;
  }
  __syncthreads();
  if (t < CC) {
    float s = ssc[t]; int cnt = 0;
    for (int j = 0; j < CC; ++j)
      cnt += (ssc[j] > s) || (ssc[j] == s && j < t);   // top_k tie-break: lower index first
    sel[t] = (cnt < KK) ? 1 : 0;
  }
  __syncthreads();
  if (t == 0) {                                        // prefix-scan -> ascending indices
    int pos = 0;
    for (int c = 0; c < CC; ++c) if (sel[c]) idx[pos++] = c;
  }
}

// ---------------- weight conversion + BN2 fold ----------------
__global__ void k_prep_w1(const float* __restrict__ w_dc1, const float* __restrict__ b_dc1,
                          const float* __restrict__ g, const float* __restrict__ b,
                          const float* __restrict__ m, const float* __restrict__ v,
                          half_t* __restrict__ w1h, float* __restrict__ alpha2,
                          float* __restrict__ beta2) {
  int i = blockIdx.x * 256 + threadIdx.x;
  if (i < CC) {
    float a = g[i] * rsqrtf(v[i] + EPSV);
    alpha2[i] = a;
    beta2[i]  = (b_dc1[i] - m[i]) * a + b[i];
  }
  if (i < CC * CATC) w1h[i] = (half_t)w_dc1[i];
}

// OIHW (512,512,3,3) -> [tap r][o][i] so GEMM2 A-tiles are contiguous in K
__global__ void k_prep_w2(const float* __restrict__ w_dc2, half_t* __restrict__ w2h) {
  int j = blockIdx.x * 256 + threadIdx.x;   // j < 9*512*512
  int r = j / (CC * CC);
  int rem = j - r * (CC * CC);
  int o = rem >> 9, c = rem & 511;
  w2h[j] = (half_t)w_dc2[(size_t)o * (CC * 9) + c * 9 + r];
}

// ---------------- build catT[25600][1024] (pixel-major), LDS-tiled transpose ----------------
__global__ __launch_bounds__(256) void k_pack_cat_t(
    const float* __restrict__ x, const float* __restrict__ scales,
    const int* __restrict__ idx, const float* __restrict__ w_cheap,
    const float* __restrict__ g1, const float* __restrict__ b1,
    const float* __restrict__ m1, const float* __restrict__ v1,
    half_t* __restrict__ catT) {
  __shared__ half_t t[32][72];
  int tid = threadIdx.x;
  int p0 = blockIdx.x * 64, c0 = blockIdx.y * 32;
  int pl = tid & 63, cq = tid >> 6;          // 4 channels per pass, 8 passes
  int p = p0 + pl;
  int y = p / WD, xcol = p - y * WD;
#pragma unroll 1
  for (int q = 0; q < 8; ++q) {
    int cl = q * 4 + cq;
    int c = c0 + cl;
    float val;
    if (c < CC) {
      val = x[(size_t)c * HWSZ + p] * scales[c];
    } else if (c < CC + KK) {
      val = x[(size_t)idx[c - CC] * HWSZ + p];
    } else {
      int j = c - CC - KK;
      const float* base = x + (size_t)idx[j] * HWSZ;
      float accv = 0.f;
#pragma unroll
      for (int ky = 0; ky < 3; ++ky) {
        int yy = y + ky - 1;
        if (yy < 0 || yy >= WD) continue;
#pragma unroll
        for (int kx = 0; kx < 3; ++kx) {
          int xq = xcol + kx - 1;
          if (xq < 0 || xq >= WD) continue;
          accv += base[yy * WD + xq] * w_cheap[j * 9 + ky * 3 + kx];
        }
      }
      float a = g1[j] * rsqrtf(v1[j] + EPSV);
      val = fmaxf((accv - m1[j]) * a + b1[j], 0.f);
    }
    t[cl][pl] = (half_t)val;
  }
  __syncthreads();
  int cl2 = tid & 31, pq = tid >> 5;
#pragma unroll 1
  for (int q = 0; q < 8; ++q) {
    int plq = q * 8 + pq;
    catT[(size_t)(p0 + plq) * CATC + c0 + cl2] = t[cl2][plq];
  }
}

// ---------------- fragment loaders (ISA 7.12.2 wave32 layouts) ----------------
__device__ __forceinline__ v16h load_a_frag(const half_t* As, int row, int lane) {
  int kb = (lane >> 4) * 8;                  // lanes 0-15: K{0-7,16-23}; 16-31: K{8-15,24-31}
  Frag f;
  f.h[0] = *(const v8h*)(As + row * LDA + kb);
  f.h[1] = *(const v8h*)(As + row * LDA + kb + 16);
  return f.v;
}
__device__ __forceinline__ v16h load_b_frag(const half_t* Bcm, int col, int lane) {
  int kb = (lane >> 4) * 8;                  // lanes 0-15: K{0-7,16-23}; 16-31: K{8-15,24-31}
  const half_t* p = Bcm + col * LDB + kb;
  Frag f;
  f.h[0] = *(const v8h*)(p);
  f.h[1] = *(const v8h*)(p + 16);
  return f.v;
}

#define WMMA_F16(a, b, c) \
  __builtin_amdgcn_wmma_f32_16x16x32_f16(false, (a), false, (b), (short)0, (c), false, false)

// ---------------- GEMM1: hT[n][512] = W1[512,1024] x catT ; TDM tile fills ----------------
__global__ __launch_bounds__(256) void k_gemm1(const half_t* __restrict__ w1h,
                                               const half_t* __restrict__ catT,
                                               const float* __restrict__ alpha2,
                                               const float* __restrict__ beta2,
                                               half_t* __restrict__ hT) {
  __shared__ half_t sA[2][BM * LDA];
  __shared__ half_t sB[2][BN * LDB];
  int tid = threadIdx.x, lane = tid & 31, w = tid >> 5;
  int tileN = blockIdx.x * BN, tileM = blockIdx.y * BM;
  int m32 = (w & 3) * 32, nb = (w >> 2) * 64;
  v8f acc[2][4] = {};
  const int KT = CATC / BK;

#ifdef HAVE_TDM
  // one TDM descriptor per tile: wave0 moves A, wave1 moves B; pad -> 40-half LDS pitch
  if (w == 0)
    tdm_load_tile(w1h + (size_t)tileM * CATC, lds_off(&sA[0][0]), CATC, CC, CATC, BM);
  else if (w == 1)
    tdm_load_tile(catT + (size_t)tileN * CATC, lds_off(&sB[0][0]), CATC, HWSZ, CATC, BN);
  if (w < 2) __builtin_amdgcn_s_wait_tensorcnt(0);
  __syncthreads();
  for (int kt = 0; kt < KT; ++kt) {
    int cur = kt & 1, nxt = cur ^ 1;
    if (kt + 1 < KT) {
      if (w == 0)
        tdm_load_tile(w1h + (size_t)tileM * CATC + (kt + 1) * BK, lds_off(&sA[nxt][0]),
                      CATC, CC, CATC, BM);
      else if (w == 1)
        tdm_load_tile(catT + (size_t)tileN * CATC + (kt + 1) * BK, lds_off(&sB[nxt][0]),
                      CATC, HWSZ, CATC, BN);
    }
    v16h a0 = load_a_frag(sA[cur], m32 + (lane & 15), lane);
    v16h a1 = load_a_frag(sA[cur], m32 + 16 + (lane & 15), lane);
#pragma unroll
    for (int j = 0; j < 4; ++j) {
      v16h b = load_b_frag(sB[cur], nb + j * 16 + (lane & 15), lane);
      acc[0][j] = WMMA_F16(a0, b, acc[0][j]);
      acc[1][j] = WMMA_F16(a1, b, acc[1][j]);
    }
    if (w < 2) __builtin_amdgcn_s_wait_tensorcnt(0);
    __syncthreads();
  }
#else
  int row = tid >> 1, seg = (tid & 1) * 16;
  const half_t* aSrc = w1h + (size_t)(tileM + row) * CATC + seg;
  const half_t* bSrc = catT + (size_t)(tileN + row) * CATC + seg;
  async_b128(aSrc, &sA[0][row * LDA + seg]);
  async_b128(aSrc + 8, &sA[0][row * LDA + seg + 8]);
  async_b128(bSrc, &sB[0][row * LDB + seg]);
  async_b128(bSrc + 8, &sB[0][row * LDB + seg + 8]);
  async_wait0();
  __syncthreads();
  for (int kt = 0; kt < KT; ++kt) {
    int cur = kt & 1, nxt = cur ^ 1;
    if (kt + 1 < KT) {
      const half_t* a = aSrc + (kt + 1) * BK;
      const half_t* b = bSrc + (kt + 1) * BK;
      async_b128(a, &sA[nxt][row * LDA + seg]);
      async_b128(a + 8, &sA[nxt][row * LDA + seg + 8]);
      async_b128(b, &sB[nxt][row * LDB + seg]);
      async_b128(b + 8, &sB[nxt][row * LDB + seg + 8]);
    }
    v16h a0 = load_a_frag(sA[cur], m32 + (lane & 15), lane);
    v16h a1 = load_a_frag(sA[cur], m32 + 16 + (lane & 15), lane);
#pragma unroll
    for (int j = 0; j < 4; ++j) {
      v16h b = load_b_frag(sB[cur], nb + j * 16 + (lane & 15), lane);
      acc[0][j] = WMMA_F16(a0, b, acc[0][j]);
      acc[1][j] = WMMA_F16(a1, b, acc[1][j]);
    }
    async_wait0();
    __syncthreads();
  }
#endif
  // epilogue: 8 contiguous output channels per lane -> one 16B store into hT[n][m]
#pragma unroll
  for (int mt = 0; mt < 2; ++mt) {
    int mrow0 = tileM + m32 + mt * 16 + (lane >> 4) * 8;
#pragma unroll
    for (int j = 0; j < 4; ++j) {
      int n = tileN + nb + j * 16 + (lane & 15);
      Frag pk;
#pragma unroll
      for (int i = 0; i < 8; ++i) {
        float vv = acc[mt][j][i] * alpha2[mrow0 + i] + beta2[mrow0 + i];
        pk.h[0][i] = (half_t)vv;
      }
      *(v8h*)(hT + (size_t)n * CC + mrow0) = pk.h[0];
    }
  }
}

// ---------------- GEMM2: implicit 3x3 over hT; async fills (halo predication) ----------------
__device__ __forceinline__ void g2_fill(half_t* As, half_t* Bs, const half_t* w2h,
                                        const half_t* hT, int tileM, int tileN,
                                        int row, int seg, int kt) {
  int cblk = kt / 9, r = kt - cblk * 9;
  int c0 = cblk * 32;
  int dy = r / 3 - 1, dx = r - (r / 3) * 3 - 1;
  const half_t* asrc = w2h + ((size_t)r * CC + tileM + row) * CC + c0 + seg;
  half_t* ad = As + row * LDA + seg;
  async_b128(asrc, ad);
  async_b128(asrc + 8, ad + 8);
  int n = tileN + row;
  int y = n / WD, xcol = n - y * WD;
  int yy = y + dy, xx = xcol + dx;
  half_t* bd = Bs + row * LDB + seg;
  if (yy >= 0 && yy < WD && xx >= 0 && xx < WD) {
    const half_t* bsrc = hT + (size_t)(yy * WD + xx) * CC + c0 + seg;
    async_b128(bsrc, bd);
    async_b128(bsrc + 8, bd + 8);
  } else {
    v8h z = {};
    *(v8h*)bd = z;
    *(v8h*)(bd + 8) = z;
  }
}

__global__ __launch_bounds__(256) void k_gemm2(const half_t* __restrict__ w2h,
                                               const half_t* __restrict__ hT,
                                               const float* __restrict__ b_dc2,
                                               float* __restrict__ out) {
  __shared__ half_t sA[2][BM * LDA];
  __shared__ half_t sB[2][BN * LDB];
  int tid = threadIdx.x, lane = tid & 31, w = tid >> 5;
  int tileN = blockIdx.x * BN, tileM = blockIdx.y * BM;
  int row = tid >> 1, seg = (tid & 1) * 16;
  int m32 = (w & 3) * 32, nb = (w >> 2) * 64;
  v8f acc[2][4] = {};
  const int KT = 16 * 9;                     // 16 channel-blocks x 9 taps

  g2_fill(sA[0], sB[0], w2h, hT, tileM, tileN, row, seg, 0);
  async_wait0();
  __syncthreads();

  for (int kt = 0; kt < KT; ++kt) {
    int cur = kt & 1, nxt = cur ^ 1;
    if (kt + 1 < KT)
      g2_fill(sA[nxt], sB[nxt], w2h, hT, tileM, tileN, row, seg, kt + 1);
    v16h a0 = load_a_frag(sA[cur], m32 + (lane & 15), lane);
    v16h a1 = load_a_frag(sA[cur], m32 + 16 + (lane & 15), lane);
#pragma unroll
    for (int j = 0; j < 4; ++j) {
      v16h b = load_b_frag(sB[cur], nb + j * 16 + (lane & 15), lane);
      acc[0][j] = WMMA_F16(a0, b, acc[0][j]);
      acc[1][j] = WMMA_F16(a1, b, acc[1][j]);
    }
    async_wait0();
    __syncthreads();
  }
#pragma unroll
  for (int mt = 0; mt < 2; ++mt) {
    int mrow0 = tileM + m32 + mt * 16 + (lane >> 4) * 8;
#pragma unroll
    for (int j = 0; j < 4; ++j) {
      int n = tileN + nb + j * 16 + (lane & 15);
#pragma unroll
      for (int i = 0; i < 8; ++i)
        out[(size_t)(mrow0 + i) * HWSZ + n] = fmaxf(acc[mt][j][i] + b_dc2[mrow0 + i], 0.f);
    }
  }
}

extern "C" void kernel_launch(void* const* d_in, const int* in_sizes, int n_in,
                              void* d_out, int out_size, void* d_ws, size_t ws_size,
                              hipStream_t stream) {
  const float* x       = (const float*)d_in[0];
  const float* w_fc1   = (const float*)d_in[1];
  const float* w_fc2   = (const float*)d_in[2];
  const float* w_cheap = (const float*)d_in[3];
  const float* bn1_g   = (const float*)d_in[4];
  const float* bn1_b   = (const float*)d_in[5];
  const float* bn1_m   = (const float*)d_in[6];
  const float* bn1_v   = (const float*)d_in[7];
  const float* w_dc1   = (const float*)d_in[8];
  const float* b_dc1   = (const float*)d_in[9];
  const float* bn2_g   = (const float*)d_in[10];
  const float* bn2_b   = (const float*)d_in[11];
  const float* bn2_m   = (const float*)d_in[12];
  const float* bn2_v   = (const float*)d_in[13];
  const float* w_dc2   = (const float*)d_in[14];
  const float* b_dc2   = (const float*)d_in[15];
  float* out = (float*)d_out;

  char* ws = (char*)d_ws;
  float* avg    = (float*)(ws);
  float* mxp    = (float*)(ws + 2048);
  float* scales = (float*)(ws + 4096);
  float* alpha2 = (float*)(ws + 6144);
  float* beta2  = (float*)(ws + 8192);
  int*   idx    = (int*)  (ws + 10240);
  half_t* w1h  = (half_t*)(ws + 16384);                 // 512*1024 halfs
  half_t* w2h  = w1h + (size_t)CC * CATC;               // 9*512*512 halfs
  half_t* catT = w2h + (size_t)9 * CC * CC;             // 25600*1024 halfs (pixel-major)
  half_t* hT   = catT + (size_t)CATC * HWSZ;            // 25600*512 halfs (pixel-major)

  k_reduce<<<CC, 256, 0, stream>>>(x, avg, mxp);
  k_mlp_topk<<<1, 512, 0, stream>>>(avg, mxp, w_fc1, w_fc2, scales, idx);
  k_prep_w1<<<(CC * CATC) / 256, 256, 0, stream>>>(w_dc1, b_dc1, bn2_g, bn2_b, bn2_m,
                                                   bn2_v, w1h, alpha2, beta2);
  k_prep_w2<<<(9 * CC * CC) / 256, 256, 0, stream>>>(w_dc2, w2h);
  dim3 gp(HWSZ / 64, CATC / 32);
  k_pack_cat_t<<<gp, 256, 0, stream>>>(x, scales, idx, w_cheap, bn1_g, bn1_b, bn1_m,
                                       bn1_v, catT);
  dim3 gg(HWSZ / BN, CC / BM);
  k_gemm1<<<gg, 256, 0, stream>>>(w1h, catT, alpha2, beta2, hT);
  k_gemm2<<<gg, 256, 0, stream>>>(w2h, hT, b_dc2, out);
}
